// EncoderLayer_88880053223871
// MI455X (gfx1250) — matrix operand order repaired and checked by reading, not compiled, not run on hardware
//
#include <hip/hip_runtime.h>

#define B_   2
#define S_   4096
#define D_   512
#define H_   8
#define DK_  64
#define DFF_ 2048

typedef unsigned short u16;
typedef __attribute__((ext_vector_type(16))) __bf16 v16bf;
typedef __attribute__((ext_vector_type(8)))  float  v8f;

static __device__ __forceinline__ u16 f2bf_bits(float f) {
  unsigned u = __builtin_bit_cast(unsigned, f);
  u += 0x7FFFu + ((u >> 16) & 1u);           // round-to-nearest-even
  return (u16)(u >> 16);
}
static __device__ __forceinline__ __bf16 f2bf(float f) {
  u16 h = f2bf_bits(f);
  return __builtin_bit_cast(__bf16, h);
}
static __device__ __forceinline__ __bf16 bfbits(u16 h) {
  return __builtin_bit_cast(__bf16, h);
}
// LDS byte offset of a shared-memory pointer (generic addr low 32 bits map to LDS)
static __device__ __forceinline__ unsigned lds_off(const void* p) {
  return (unsigned)(size_t)p;
}

// ---------------------------------------------------------------------------
// Tiled bf16 WMMA GEMM: C[M,N] = act(A[M,K] @ B[K,N] + bias)
// 256 threads = 8 waves arranged 2(M)x4(N); BM=BN=128, BK=64.
// Double-buffered LDS pipeline: fp32 operands prefetched to registers,
// bf16 A operand staged with GLOBAL_LOAD_ASYNC_TO_LDS_B128 (ASYNCcnt).
// EPI: 0 = none, 1 = +bias, 2 = +bias, ReLU
// ---------------------------------------------------------------------------
template <bool A_BF16, bool OUT_BF16, int EPI>
__global__ __launch_bounds__(256) void gemm_wmma(
    const void* __restrict__ Ap, const float* __restrict__ Bw,
    const float* __restrict__ bias, void* __restrict__ Cp,
    int M, int N, int K)
{
  constexpr int BM = 128, BN = 128, BK = 64, LD = BK + 8;
  __shared__ __bf16 sA[2][BM * LD];   // sA[buf][m][k]
  __shared__ __bf16 sB[2][BN * LD];   // sB[buf][n][k]  (B transposed)

  const int tid  = threadIdx.x;
  const int lane = tid & 31, wave = tid >> 5;
  const int hlf  = lane >> 4, l16 = lane & 15;
  const int wm   = (wave >> 2) * 64, wn = (wave & 3) * 32;
  const int mtile = blockIdx.y * BM, ntile = blockIdx.x * BN;

  const float* Af = (const float*)Ap;
  const u16*   Ab = (const u16*)Ap;

  const int rsubA = tid >> 4, kcA = (tid & 15) * 4;   // A: 16 rows/pass, 4 cols
  const int rsubB = tid >> 5, ncB = (tid & 31) * 4;   // B: 8 rows/pass, 4 cols

  v8f acc[4][2];
#pragma unroll
  for (int i = 0; i < 4; i++)
#pragma unroll
    for (int j = 0; j < 2; j++) acc[i][j] = {};

  float4 aReg[8], bReg[8];

  auto fetchA = [&](int kt) {
#pragma unroll
    for (int p = 0; p < 8; p++)
      aReg[p] = *reinterpret_cast<const float4*>(
          Af + (size_t)(mtile + p * 16 + rsubA) * K + kt + kcA);
  };
  auto storeA = [&](int buf) {
#pragma unroll
    for (int p = 0; p < 8; p++) {
      __bf16* d = &sA[buf][(p * 16 + rsubA) * LD + kcA];
      d[0] = f2bf(aReg[p].x); d[1] = f2bf(aReg[p].y);
      d[2] = f2bf(aReg[p].z); d[3] = f2bf(aReg[p].w);
    }
  };
  // bf16 A: async DMA straight into LDS, 16B per lane per issue
  auto asyncA = [&](int kt, int buf) {
#pragma unroll
    for (int p = 0; p < 4; p++) {
      const int c = p * 256 + tid;           // 1024 chunks = 128 rows x 8
      const int row = c >> 3, ch = c & 7;
      unsigned lptr = lds_off(&sA[buf][row * LD + ch * 8]);
      unsigned goff = (unsigned)(((size_t)(mtile + row) * K + kt + ch * 8) * sizeof(u16));
      asm volatile("global_load_async_to_lds_b128 %0, %1, %2"
                   :: "v"(lptr), "v"(goff),
                      "s"((unsigned long long)(size_t)Ab)
                   : "memory");
    }
  };
  auto fetchB = [&](int kt) {
#pragma unroll
    for (int p = 0; p < 8; p++)
      bReg[p] = *reinterpret_cast<const float4*>(
          Bw + (size_t)(kt + p * 8 + rsubB) * N + ntile + ncB);
  };
  auto storeB = [&](int buf) {
#pragma unroll
    for (int p = 0; p < 8; p++) {
      const int r = p * 8 + rsubB;
      sB[buf][(ncB + 0) * LD + r] = f2bf(bReg[p].x);
      sB[buf][(ncB + 1) * LD + r] = f2bf(bReg[p].y);
      sB[buf][(ncB + 2) * LD + r] = f2bf(bReg[p].z);
      sB[buf][(ncB + 3) * LD + r] = f2bf(bReg[p].w);
    }
  };
  auto compute = [&](int buf) {
#pragma unroll
    for (int kk = 0; kk < 2; kk++) {         // two 32-deep WMMA steps
      v16bf afr[4], bfr[2];
#pragma unroll
      for (int mi = 0; mi < 4; mi++) {
        const __bf16* p = &sA[buf][(wm + mi * 16 + l16) * LD + kk * 32 + hlf * 8];
#pragma unroll
        for (int e = 0; e < 8; e++) { afr[mi][e] = p[e]; afr[mi][8 + e] = p[16 + e]; }
      }
#pragma unroll
      for (int ni = 0; ni < 2; ni++) {
        const __bf16* p = &sB[buf][(wn + ni * 16 + l16) * LD + kk * 32 + hlf * 8];
#pragma unroll
        for (int e = 0; e < 8; e++) { bfr[ni][e] = p[e]; bfr[ni][8 + e] = p[16 + e]; }
      }
#pragma unroll
      for (int mi = 0; mi < 4; mi++)
#pragma unroll
        for (int ni = 0; ni < 2; ni++)
          acc[mi][ni] = __builtin_amdgcn_wmma_f32_16x16x32_bf16(
              false, afr[mi], false, bfr[ni], (short)0, acc[mi][ni], false, false);
    }
  };

  // --- pipelined main loop (double-buffered LDS) ---
  const int nT = K / BK;
  if constexpr (A_BF16) { asyncA(0, 0); } else { fetchA(0); storeA(0); }
  fetchB(0); storeB(0);
  if constexpr (A_BF16) asm volatile("s_wait_asynccnt 0" ::: "memory");
  __syncthreads();

  for (int t = 0; t < nT; t++) {
    const int cur = t & 1, nxt = cur ^ 1;
    const bool more = (t + 1) < nT;
    if (more) {                               // issue next tile's loads first
      if constexpr (A_BF16) asyncA((t + 1) * BK, nxt);
      else                  fetchA((t + 1) * BK);
      fetchB((t + 1) * BK);
    }
    compute(cur);                             // WMMAs overlap in-flight loads
    if (more) {
      if constexpr (!A_BF16) storeA(nxt);
      storeB(nxt);
      if constexpr (A_BF16) asm volatile("s_wait_asynccnt 0" ::: "memory");
    }
    __syncthreads();
  }

  // --- epilogue ---
#pragma unroll
  for (int mi = 0; mi < 4; mi++)
#pragma unroll
    for (int ni = 0; ni < 2; ni++) {
      const int col   = ntile + wn + ni * 16 + l16;
      const int rbase = mtile + wm + mi * 16 + hlf * 8;
      float bv = (EPI > 0) ? bias[col] : 0.0f;
#pragma unroll
      for (int e = 0; e < 8; e++) {
        float val = acc[mi][ni][e] + bv;
        if (EPI == 2) val = fmaxf(val, 0.0f);
        const size_t oi = (size_t)(rbase + e) * N + col;
        if constexpr (OUT_BF16) ((u16*)Cp)[oi] = f2bf_bits(val);
        else                    ((float*)Cp)[oi] = val;
      }
    }
}

// ---------------------------------------------------------------------------
// Flash attention: per (b,h), 128 queries per block (16 rows per wave),
// streaming 64-key tiles with online softmax. Q/K/V are bf16 in [B,S,D].
// K tiles staged via async-to-LDS DMA; V transposed through VALU.
// ---------------------------------------------------------------------------
__global__ __launch_bounds__(256) void attn_wmma(
    const u16* __restrict__ qg, const u16* __restrict__ kg,
    const u16* __restrict__ vg, const int* __restrict__ mask,
    float* __restrict__ ctx)
{
  constexpr int LD = 72;
  __shared__ __bf16 sK[64 * LD];          // sK[j][d]
  __shared__ __bf16 sV[64 * LD];          // sV[d][j] = V[j][d]
  __shared__ __bf16 sP[8 * 16 * LD];      // per-wave P scratch (16 x 64)

  const int tid = threadIdx.x, lane = tid & 31, wave = tid >> 5;
  const int hlf = lane >> 4, l16 = lane & 15;
  const int bh = blockIdx.y, b = bh / H_, h = bh % H_;
  const int qrow0 = blockIdx.x * 128 + wave * 16;

  const u16* kbase = kg + (size_t)b * S_ * D_ + h * DK_;

  // preload this wave's 16x64 Q block as two A-fragments (K=32 each)
  v16bf qf[2];
  {
    const u16* qp = qg + (size_t)(b * S_ + qrow0 + l16) * D_ + h * DK_;
#pragma unroll
    for (int kk = 0; kk < 2; kk++)
#pragma unroll
      for (int e = 0; e < 8; e++) {
        qf[kk][e]     = bfbits(qp[kk * 32 + hlf * 8 + e]);
        qf[kk][8 + e] = bfbits(qp[kk * 32 + 16 + hlf * 8 + e]);
      }
  }

  v8f cacc[4];
#pragma unroll
  for (int i = 0; i < 4; i++) cacc[i] = {};
  float rmax[8], rsum[8];
#pragma unroll
  for (int e = 0; e < 8; e++) { rmax[e] = -1e30f; rsum[e] = 0.0f; }

  for (int jt = 0; jt < S_; jt += 64) {
    // --- async DMA: K tile (64 x 64 bf16) straight into LDS ---
#pragma unroll
    for (int p = 0; p < 2; p++) {
      const int c = p * 256 + tid;         // 512 chunks = 64 rows x 8
      const int j = c >> 3, ch = c & 7;
      unsigned lptr = lds_off(&sK[j * LD + ch * 8]);
      unsigned goff = (unsigned)(((size_t)(jt + j) * D_ + ch * 8) * sizeof(u16));
      asm volatile("global_load_async_to_lds_b128 %0, %1, %2"
                   :: "v"(lptr), "v"(goff),
                      "s"((unsigned long long)(size_t)kbase)
                   : "memory");
    }
    // --- stage V tile transposed (needs lane shuffle, stays on VALU) ---
    {
      const int rsub = tid >> 4;           // 0..15
      const int dc   = (tid & 15) * 4;     // 0..60
#pragma unroll
      for (int p = 0; p < 4; p++) {
        const int j = p * 16 + rsub;
        const u16* vs = vg + (size_t)(b * S_ + jt + j) * D_ + h * DK_ + dc;
        sV[(dc + 0) * LD + j] = bfbits(vs[0]);
        sV[(dc + 1) * LD + j] = bfbits(vs[1]);
        sV[(dc + 2) * LD + j] = bfbits(vs[2]);
        sV[(dc + 3) * LD + j] = bfbits(vs[3]);
      }
    }
    asm volatile("s_wait_asynccnt 0" ::: "memory");
    __syncthreads();

    // scores S = (Q K^T)/8, masked
    float svv[4][8];
#pragma unroll
    for (int jn = 0; jn < 4; jn++) {
      v16bf bk[2];
      const __bf16* kp = &sK[(jn * 16 + l16) * LD];
#pragma unroll
      for (int kk = 0; kk < 2; kk++)
#pragma unroll
        for (int e = 0; e < 8; e++) { bk[kk][e] = kp[kk * 32 + hlf * 8 + e]; bk[kk][8 + e] = kp[kk * 32 + 16 + hlf * 8 + e]; }
      v8f sf = {};
      sf = __builtin_amdgcn_wmma_f32_16x16x32_bf16(false, qf[0], false, bk[0], (short)0, sf, false, false);
      sf = __builtin_amdgcn_wmma_f32_16x16x32_bf16(false, qf[1], false, bk[1], (short)0, sf, false, false);
      const int mz = mask[b * S_ + jt + jn * 16 + l16];
#pragma unroll
      for (int e = 0; e < 8; e++) {
        float t = sf[e] * 0.125f;
        svv[jn][e] = (mz == 0) ? -1e9f : t;
      }
    }

    // online softmax: row max across the 16-lane half, rescale accumulators
    float sc[8];
#pragma unroll
    for (int e = 0; e < 8; e++) {
      float t = fmaxf(fmaxf(svv[0][e], svv[1][e]), fmaxf(svv[2][e], svv[3][e]));
      t = fmaxf(t, __shfl_xor(t, 1, 32));
      t = fmaxf(t, __shfl_xor(t, 2, 32));
      t = fmaxf(t, __shfl_xor(t, 4, 32));
      t = fmaxf(t, __shfl_xor(t, 8, 32));
      const float nm = fmaxf(rmax[e], t);
      sc[e]   = __expf(rmax[e] - nm);
      rmax[e] = nm;
      rsum[e] *= sc[e];
    }
#pragma unroll
    for (int dn = 0; dn < 4; dn++)
#pragma unroll
      for (int e = 0; e < 8; e++) cacc[dn][e] *= sc[e];

    // P = exp(S - max): accumulate row sums, store bf16 P to per-wave LDS
    __bf16* pp = &sP[wave * 16 * LD];
    float ps[8];
#pragma unroll
    for (int e = 0; e < 8; e++) ps[e] = 0.0f;
#pragma unroll
    for (int jn = 0; jn < 4; jn++)
#pragma unroll
      for (int e = 0; e < 8; e++) {
        const float pv = __expf(svv[jn][e] - rmax[e]);
        ps[e] += pv;
        pp[(e + hlf * 8) * LD + jn * 16 + l16] = f2bf(pv);
      }
#pragma unroll
    for (int e = 0; e < 8; e++) {
      float t = ps[e];
      t += __shfl_xor(t, 1, 32);
      t += __shfl_xor(t, 2, 32);
      t += __shfl_xor(t, 4, 32);
      t += __shfl_xor(t, 8, 32);
      rsum[e] += t;
    }

    // ctx += P @ V  (P reloaded from same-wave LDS scratch; DS ops are
    // in-order within a wave so no barrier is needed for the round-trip)
    v16bf afr[2];
    {
      const __bf16* pr = pp + l16 * LD;
#pragma unroll
      for (int kk = 0; kk < 2; kk++)
#pragma unroll
        for (int e = 0; e < 8; e++) {
          afr[kk][e]     = pr[kk * 32 + hlf * 8 + e];
          afr[kk][8 + e] = pr[kk * 32 + 16 + hlf * 8 + e];
        }
    }
#pragma unroll
    for (int dn = 0; dn < 4; dn++) {
      v16bf bv[2];
      const __bf16* vp = &sV[(dn * 16 + l16) * LD];
#pragma unroll
      for (int kk = 0; kk < 2; kk++)
#pragma unroll
        for (int e = 0; e < 8; e++) { bv[kk][e] = vp[kk * 32 + hlf * 8 + e]; bv[kk][8 + e] = vp[kk * 32 + 16 + hlf * 8 + e]; }
      cacc[dn] = __builtin_amdgcn_wmma_f32_16x16x32_bf16(false, afr[0], false, bv[0], (short)0, cacc[dn], false, false);
      cacc[dn] = __builtin_amdgcn_wmma_f32_16x16x32_bf16(false, afr[1], false, bv[1], (short)0, cacc[dn], false, false);
    }
    __syncthreads();
  }

  // normalize and write fp32 ctx (heads re-merged into [B,S,D])
#pragma unroll
  for (int dn = 0; dn < 4; dn++)
#pragma unroll
    for (int e = 0; e < 8; e++) {
      const int row = qrow0 + e + hlf * 8;
      const int col = h * DK_ + dn * 16 + l16;
      ctx[(size_t)(b * S_ + row) * D_ + col] = cacc[dn][e] / rsum[e];
    }
}

// ---------------------------------------------------------------------------
// out = LayerNorm(x + y) * g + b    (one row of D=512 per 256-thread block)
// ---------------------------------------------------------------------------
__global__ __launch_bounds__(256) void ln_res(
    const float* __restrict__ x, const float* __restrict__ y,
    const float* __restrict__ g, const float* __restrict__ be,
    float* __restrict__ out)
{
  const int row = blockIdx.x, tid = threadIdx.x;
  const int lane = tid & 31, wave = tid >> 5;
  const float* xr = x + (size_t)row * D_;
  const float* yr = y + (size_t)row * D_;
  const float s0 = xr[tid] + yr[tid];
  const float s1 = xr[tid + 256] + yr[tid + 256];
  float sum = s0 + s1;
  float sq  = s0 * s0 + s1 * s1;
  for (int m = 1; m <= 16; m <<= 1) {
    sum += __shfl_xor(sum, m, 32);
    sq  += __shfl_xor(sq,  m, 32);
  }
  __shared__ float w1s[8], w2s[8];
  if (lane == 0) { w1s[wave] = sum; w2s[wave] = sq; }
  __syncthreads();
  sum = 0.0f; sq = 0.0f;
#pragma unroll
  for (int i = 0; i < 8; i++) { sum += w1s[i]; sq += w2s[i]; }
  const float mu  = sum * (1.0f / D_);
  const float var = sq * (1.0f / D_) - mu * mu;
  const float r   = rsqrtf(var + 1e-5f);
  out[(size_t)row * D_ + tid]       = (s0 - mu) * r * g[tid]       + be[tid];
  out[(size_t)row * D_ + tid + 256] = (s1 - mu) * r * g[tid + 256] + be[tid + 256];
}

// ---------------------------------------------------------------------------
extern "C" void kernel_launch(void* const* d_in, const int* in_sizes, int n_in,
                              void* d_out, int out_size, void* d_ws, size_t ws_size,
                              hipStream_t stream)
{
  (void)in_sizes; (void)n_in; (void)out_size; (void)ws_size;

  const float* x    = (const float*)d_in[0];
  const int*   mask = (const int*)  d_in[1];
  const float* wq   = (const float*)d_in[2];
  const float* wk   = (const float*)d_in[3];
  const float* wv   = (const float*)d_in[4];
  const float* wo   = (const float*)d_in[5];
  const float* w1   = (const float*)d_in[6];
  const float* b1   = (const float*)d_in[7];
  const float* w2   = (const float*)d_in[8];
  const float* b2   = (const float*)d_in[9];
  const float* g1   = (const float*)d_in[10];
  const float* be1  = (const float*)d_in[11];
  const float* g2   = (const float*)d_in[12];
  const float* be2  = (const float*)d_in[13];

  const size_t BS = (size_t)B_ * S_;     // 8192 rows
  char* p = (char*)d_ws;
  auto take = [&](size_t bytes) -> void* {
    void* r = (void*)p;
    p += (bytes + 255) & ~(size_t)255;
    return r;
  };
  u16*   qb   = (u16*)  take(BS * D_ * 2);
  u16*   kb   = (u16*)  take(BS * D_ * 2);
  u16*   vb   = (u16*)  take(BS * D_ * 2);
  float* ctxb = (float*)take(BS * D_ * 4);
  float* yb   = (float*)take(BS * D_ * 4);
  float* x1b  = (float*)take(BS * D_ * 4);
  u16*   hb   = (u16*)  take(BS * DFF_ * 2);
  float* zb   = (float*)take(BS * D_ * 4);

  const dim3 blk(256);
  const dim3 gP(D_ / 128, (unsigned)(BS / 128));       // 4 x 64

  // QKV projections (fp32 in, bf16 out for the attention WMMAs)
  gemm_wmma<false, true, 0><<<gP, blk, 0, stream>>>(x, wq, nullptr, qb, (int)BS, D_, D_);
  gemm_wmma<false, true, 0><<<gP, blk, 0, stream>>>(x, wk, nullptr, kb, (int)BS, D_, D_);
  gemm_wmma<false, true, 0><<<gP, blk, 0, stream>>>(x, wv, nullptr, vb, (int)BS, D_, D_);

  // flash attention -> fp32 ctx [B,S,D]
  attn_wmma<<<dim3(S_ / 128, B_ * H_), blk, 0, stream>>>(qb, kb, vb, mask, ctxb);

  // output projection, residual + LN1
  gemm_wmma<false, false, 0><<<gP, blk, 0, stream>>>(ctxb, wo, nullptr, yb, (int)BS, D_, D_);
  ln_res<<<dim3((unsigned)BS), blk, 0, stream>>>(x, yb, g1, be1, x1b);

  // FFN: relu(x1 @ w1 + b1) @ w2 + b2, residual + LN2 -> out
  gemm_wmma<false, true, 2><<<dim3(DFF_ / 128, (unsigned)(BS / 128)), blk, 0, stream>>>(
      x1b, w1, b1, hb, (int)BS, DFF_, D_);
  gemm_wmma<true, false, 1><<<gP, blk, 0, stream>>>(hb, w2, b2, zb, (int)BS, D_, DFF_);
  ln_res<<<dim3((unsigned)BS), blk, 0, stream>>>(x1b, zb, g2, be2, (float*)d_out);
}
